// SwinTransformerBlock_81819126989606
// MI455X (gfx1250) — matrix operand order repaired and checked by reading, not compiled, not run on hardware
//
#include <hip/hip_runtime.h>
#include <hip/hip_bf16.h>
#include <math.h>

typedef __bf16 bf16;
typedef __attribute__((ext_vector_type(16))) __bf16 bf16x16;
typedef __attribute__((ext_vector_type(8)))  __bf16 bf16x8;
typedef __attribute__((ext_vector_type(8)))  float  f32x8;

constexpr int BATCH = 32;
constexpr int HW    = 56;      // H == W
constexpr int CDIM  = 128;
constexpr int HEADS = 4;
constexpr int HDIM  = 32;      // head dim
constexpr int WS7   = 7;
constexpr int SH    = 3;       // shift
constexpr int NW    = 49;      // tokens per window
constexpr int NP    = 64;      // padded window rows
constexpr int MLPD  = 512;
constexpr int NWIN  = 64;      // windows per image (8x8)
constexpr int BW    = BATCH * NWIN;       // 2048 windows total
constexpr int NTOK  = BATCH * HW * HW;    // 100352 tokens
constexpr float SCALE_Q = 0.17677669529663687f;  // 1/sqrt(32)
constexpr float EPS_LN  = 1e-5f;

__device__ __forceinline__ int lane_id() { return threadIdx.x & 31; }
__device__ __forceinline__ int wave_id() { return threadIdx.x >> 5; }

__device__ __forceinline__ float wred_sum(float v) {
#pragma unroll
  for (int off = 16; off > 0; off >>= 1) v += __shfl_xor(v, off, 32);
  return v;
}
__device__ __forceinline__ float wred_max(float v) {
#pragma unroll
  for (int off = 16; off > 0; off >>= 1) v = fmaxf(v, __shfl_xor(v, off, 32));
  return v;
}

// ---- WMMA fragment loaders (CDNA5 ISA 7.12.2 layouts, wave32) ----
// A matrix 16x32 bf16: lane = (M | half<<4); elem e<8 -> K = e + 8*half,
// elem e>=8 -> K = 16 + (e-8) + 8*half. Tile is row-major [16][ldk].
__device__ __forceinline__ bf16x16 load_fragA(const bf16* tile, int ldk) {
  int l = lane_id();
  const bf16* p = tile + (l & 15) * ldk + (l >> 4) * 8;
  bf16x8 lo = *(const bf16x8*)p;
  bf16x8 hi = *(const bf16x8*)(p + 16);
  bf16x16 r;
#pragma unroll
  for (int i = 0; i < 8; ++i) { r[i] = lo[i]; r[i + 8] = hi[i]; }
  return r;
}
// B matrix 32x16 bf16: lane = (N | half<<4); elem e<8 -> K = e + 8*half,
// elem e>=8 -> K = 16 + (e-8) + 8*half.
// Tile stored transposed row-major [n][ldk] (K contiguous).
__device__ __forceinline__ bf16x16 load_fragB(const bf16* tile, int ldk) {
  int l = lane_id();
  const bf16* p = tile + (l & 15) * ldk + (l >> 4) * 8;
  bf16x8 lo = *(const bf16x8*)p;
  bf16x8 hi = *(const bf16x8*)(p + 16);
  bf16x16 r;
#pragma unroll
  for (int i = 0; i < 8; ++i) { r[i] = lo[i]; r[i + 8] = hi[i]; }
  return r;
}

__device__ __forceinline__ f32x8 wmma_bf16(bf16x16 a, bf16x16 b, f32x8 c) {
  return __builtin_amdgcn_wmma_f32_16x16x32_bf16(false, a, false, b, (short)0, c,
                                                 false, false);
}

// Prefetch one 16B line of an upcoming fragment (lowers to global_prefetch_b8).
__device__ __forceinline__ void pf(const bf16* p) {
  __builtin_prefetch((const void*)p, 0, 3);
}

// ------------------------- prep kernels -------------------------
// w[K][Ncol] (f32) -> wT[Ncol][K] (bf16)
__global__ void __launch_bounds__(256)
prep_transpose(const float* __restrict__ w, bf16* __restrict__ wT, int K, int Ncol) {
  int total = K * Ncol;
  for (int idx = blockIdx.x * blockDim.x + threadIdx.x; idx < total;
       idx += gridDim.x * blockDim.x) {
    int n = idx / K, k = idx - n * K;
    wT[idx] = (bf16)w[k * Ncol + n];
  }
}

__device__ __forceinline__ int region56(int p) { return p < 49 ? 0 : (p < 53 ? 1 : 2); }

__global__ void __launch_bounds__(256)
prep_bias_mask(const float* __restrict__ table, float* __restrict__ biasT,
               float* __restrict__ maskT) {
  int tid = blockIdx.x * blockDim.x + threadIdx.x;
  int stride = gridDim.x * blockDim.x;
  // biasT[head][mq][nk] padded to 64x64 (pads = 0)
  for (int idx = tid; idx < HEADS * NP * NP; idx += stride) {
    int h = idx >> 12, m = (idx >> 6) & 63, n = idx & 63;
    float v = 0.f;
    if (m < NW && n < NW) {
      int iq = m / WS7, jq = m % WS7, ik = n / WS7, jk = n % WS7;
      int ridx = (iq - ik + WS7 - 1) * (2 * WS7 - 1) + (jq - jk + WS7 - 1);
      v = table[ridx * HEADS + h];
    }
    biasT[idx] = v;
  }
  // maskT[win][mq][nk]: shift-window mask; padded key cols get -1e9
  for (int idx = tid; idx < NWIN * NP * NP; idx += stride) {
    int ww = idx >> 12, m = (idx >> 6) & 63, n = idx & 63;
    float v;
    if (n >= NW) v = -1e9f;
    else if (m >= NW) v = 0.f;
    else {
      int wh = ww >> 3, wx = ww & 7;
      int hq = wh * WS7 + m / WS7, wq = wx * WS7 + m % WS7;
      int hk = wh * WS7 + n / WS7, wk = wx * WS7 + n % WS7;
      int rq = 3 * region56(hq) + region56(wq);
      int rk = 3 * region56(hk) + region56(wk);
      v = (rq != rk) ? -100.f : 0.f;
    }
    maskT[idx] = v;
  }
}

// ---------------- K1: LN1 + window partition + QKV GEMM ----------------
__global__ void __launch_bounds__(256)
k1_ln_qkv(const float* __restrict__ x, const float* __restrict__ g1,
          const float* __restrict__ b1, const bf16* __restrict__ qkv_wT,
          const float* __restrict__ qkv_b, bf16* __restrict__ qbuf,
          bf16* __restrict__ kbuf, bf16* __restrict__ vT) {
  __shared__ __align__(16) bf16 Ah[NP * CDIM];  // 16 KB
  int wi = blockIdx.x;
  int b = wi >> 6, ww = wi & 63, wh = ww >> 3, wx = ww & 7;
  int lane = lane_id(), wave = wave_id();

  // LN over 128 channels; roll(-SH) + window partition fused into the gather
  for (int r = wave; r < NW; r += 8) {
    int i = r / WS7, j = r % WS7;
    int hr = (wh * WS7 + i + SH) % HW;
    int wc = (wx * WS7 + j + SH) % HW;
    const float4 xv = *(const float4*)(x + (((size_t)b * HW + hr) * HW + wc) * CDIM + lane * 4);
    float mu = wred_sum(xv.x + xv.y + xv.z + xv.w) * (1.f / CDIM);
    float d0 = xv.x - mu, d1 = xv.y - mu, d2 = xv.z - mu, d3 = xv.w - mu;
    float var = wred_sum(d0 * d0 + d1 * d1 + d2 * d2 + d3 * d3) * (1.f / CDIM);
    float rs = rsqrtf(var + EPS_LN);
    const float4 gv = *(const float4*)(g1 + lane * 4);
    const float4 bv = *(const float4*)(b1 + lane * 4);
    bf16* dst = Ah + r * CDIM + lane * 4;
    dst[0] = (bf16)(d0 * rs * gv.x + bv.x);
    dst[1] = (bf16)(d1 * rs * gv.y + bv.y);
    dst[2] = (bf16)(d2 * rs * gv.z + bv.z);
    dst[3] = (bf16)(d3 * rs * gv.w + bv.w);
  }
  for (int p = threadIdx.x; p < (NP - NW) * CDIM; p += 256) Ah[NW * CDIM + p] = (bf16)0.f;
  __syncthreads();

  // 64x384 = 4 Mtiles x 24 Ntiles, K=128 (4 WMMA steps)
  for (int t = wave; t < 96; t += 8) {
    int mt = t / 24, nt = t % 24;
    f32x8 acc = {};
    const bf16* Ab = Ah + mt * 16 * CDIM;
    const bf16* Bb = qkv_wT + nt * 16 * CDIM;
#pragma unroll
    for (int ks = 0; ks < 4; ++ks) {
      if (ks < 3) pf(Bb + (ks + 1) * 32 + (lane & 15) * CDIM);
      acc = wmma_bf16(load_fragA(Ab + ks * 32, CDIM), load_fragB(Bb + ks * 32, CDIM), acc);
    }
    // Tile-uniform destination select: `which` is constant across the tile
    // because q/k/v column boundaries (128,256) are multiples of 16.
    int col = nt * 16 + (lane & 15);
    int which = (nt * 16) >> 7;          // 0=q, 1=k, 2=v (wave-uniform)
    int hcol = col & 127;
    int head = hcol >> 5, d = hcol & 31;
    int rb = (lane >> 4) * 8;
    float bias = qkv_b[col];
    size_t wiH = (size_t)wi * HEADS + head;
    if (which < 2) {
      // q and k share the [row][d] layout; q additionally scaled.
      bf16* dstq = (which == 0 ? qbuf : kbuf) + (wiH * NP) * HDIM + d;
      float sc = (which == 0) ? SCALE_Q : 1.f;
#pragma unroll
      for (int i = 0; i < 8; ++i) {
        int m = mt * 16 + rb + i;
        float v = (m < NW) ? (acc[i] + bias) * sc : 0.f;
        dstq[(size_t)m * HDIM] = (bf16)v;
      }
    } else {
      bf16* dstv = vT + (wiH * HDIM + d) * NP;  // transposed V: [d][keyrow]
#pragma unroll
      for (int i = 0; i < 8; ++i) {
        int m = mt * 16 + rb + i;
        float v = (m < NW) ? (acc[i] + bias) : 0.f;
        dstv[m] = (bf16)v;
      }
    }
  }
}

// ---------------- K2: attention per (window, head) ----------------
__global__ void __launch_bounds__(256)
k2_attn(const bf16* __restrict__ qbuf, const bf16* __restrict__ kbuf,
        const bf16* __restrict__ vT, const float* __restrict__ biasT,
        const float* __restrict__ maskT, bf16* __restrict__ owin) {
  __shared__ __align__(16) float S[NP * NP];  // 16 KB scores
  __shared__ __align__(16) bf16 P[NP * NP];   // 8 KB probs
  int bid = blockIdx.x;                       // wi*HEADS + head
  int wi = bid >> 2, head = bid & 3, ww = wi & 63;
  int lane = lane_id(), wave = wave_id();

  // scores = q @ k^T : 4x4 tiles, K=32 (single WMMA each)
  for (int t = wave; t < 16; t += 8) {
    int mt = t >> 2, nt = t & 3;
    f32x8 acc = {};
    bf16x16 a = load_fragA(qbuf + ((size_t)bid * NP + mt * 16) * HDIM, HDIM);
    bf16x16 bb = load_fragB(kbuf + ((size_t)bid * NP + nt * 16) * HDIM, HDIM);
    acc = wmma_bf16(a, bb, acc);
    int col = nt * 16 + (lane & 15);
    int rb = (lane >> 4) * 8;
#pragma unroll
    for (int i = 0; i < 8; ++i) {
      int m = mt * 16 + rb + i;
      S[m * NP + col] = acc[i] + biasT[((size_t)head * NP + m) * NP + col]
                               + maskT[((size_t)ww * NP + m) * NP + col];
    }
  }
  __syncthreads();
  // row softmax (padded key cols already at -1e9 -> exp = 0)
  for (int r = wave; r < NP; r += 8) {
    float s0 = S[r * NP + lane], s1 = S[r * NP + lane + 32];
    float mx = wred_max(fmaxf(s0, s1));
    float e0 = __expf(s0 - mx), e1 = __expf(s1 - mx);
    float inv = 1.f / wred_sum(e0 + e1);
    P[r * NP + lane]      = (bf16)(e0 * inv);
    P[r * NP + lane + 32] = (bf16)(e1 * inv);
  }
  __syncthreads();
  // o = P @ V : 4x2 tiles, K=64 (2 WMMA steps); one tile per wave
  {
    int mt = wave >> 1, nt = wave & 1;
    f32x8 acc = {};
#pragma unroll
    for (int ks = 0; ks < 2; ++ks)
      acc = wmma_bf16(load_fragA(P + mt * 16 * NP + ks * 32, NP),
                      load_fragB(vT + ((size_t)bid * HDIM + nt * 16) * NP + ks * 32, NP), acc);
    int col = nt * 16 + (lane & 15);
    int rb = (lane >> 4) * 8;
#pragma unroll
    for (int i = 0; i < 8; ++i) {
      int m = mt * 16 + rb + i;
      owin[((size_t)wi * NP + m) * CDIM + head * HDIM + col] =
          (m < NW) ? (bf16)acc[i] : (bf16)0.f;
    }
  }
}

// ---------------- K3: proj + window reverse + roll + residual ----------------
__global__ void __launch_bounds__(256)
k3_proj(const float* __restrict__ x, const bf16* __restrict__ owin,
        const bf16* __restrict__ proj_wT, const float* __restrict__ proj_b,
        float* __restrict__ out) {
  int wi = blockIdx.x;
  int b = wi >> 6, ww = wi & 63, wh = ww >> 3, wx = ww & 7;
  int lane = lane_id(), wave = wave_id();
  for (int t = wave; t < 32; t += 8) {
    int mt = t >> 3, nt = t & 7;
    f32x8 acc = {};
    const bf16* Ab = owin + ((size_t)wi * NP + mt * 16) * CDIM;
    const bf16* Bb = proj_wT + nt * 16 * CDIM;
#pragma unroll
    for (int ks = 0; ks < 4; ++ks) {
      if (ks < 3) {
        pf(Ab + (ks + 1) * 32 + (lane & 15) * CDIM);
        pf(Bb + (ks + 1) * 32 + (lane & 15) * CDIM);
      }
      acc = wmma_bf16(load_fragA(Ab + ks * 32, CDIM), load_fragB(Bb + ks * 32, CDIM), acc);
    }
    int col = nt * 16 + (lane & 15);
    int rb = (lane >> 4) * 8;
    float pb = proj_b[col];
#pragma unroll
    for (int i = 0; i < 8; ++i) {
      int m = mt * 16 + rb + i;
      if (m < NW) {
        int ii = m / WS7, jj = m % WS7;
        int hr = (wh * WS7 + ii + SH) % HW;  // reverse(+SH) of the K1 gather
        int wc = (wx * WS7 + jj + SH) % HW;
        size_t pos = (((size_t)b * HW + hr) * HW + wc) * CDIM + col;
        out[pos] = x[pos] + acc[i] + pb;     // x1 residual into d_out
      }
    }
  }
}

// ---------------- K4: LN2 + fc1 + GELU ----------------
__global__ void __launch_bounds__(256)
k4_ln_fc1(const float* __restrict__ x1, const float* __restrict__ g2,
          const float* __restrict__ b2, const bf16* __restrict__ fc1_wT,
          const float* __restrict__ fc1_b, bf16* __restrict__ hmlp) {
  __shared__ __align__(16) bf16 Ah[NP * CDIM];
  size_t tok0 = (size_t)blockIdx.x * NP;   // 1568 blocks * 64 tokens == 100352 exactly
  int lane = lane_id(), wave = wave_id();
  for (int r = wave; r < NP; r += 8) {
    const float4 xv = *(const float4*)(x1 + (tok0 + r) * CDIM + lane * 4);
    float mu = wred_sum(xv.x + xv.y + xv.z + xv.w) * (1.f / CDIM);
    float d0 = xv.x - mu, d1 = xv.y - mu, d2 = xv.z - mu, d3 = xv.w - mu;
    float var = wred_sum(d0 * d0 + d1 * d1 + d2 * d2 + d3 * d3) * (1.f / CDIM);
    float rs = rsqrtf(var + EPS_LN);
    const float4 gv = *(const float4*)(g2 + lane * 4);
    const float4 bv = *(const float4*)(b2 + lane * 4);
    bf16* dst = Ah + r * CDIM + lane * 4;
    dst[0] = (bf16)(d0 * rs * gv.x + bv.x);
    dst[1] = (bf16)(d1 * rs * gv.y + bv.y);
    dst[2] = (bf16)(d2 * rs * gv.z + bv.z);
    dst[3] = (bf16)(d3 * rs * gv.w + bv.w);
  }
  __syncthreads();
  // 64x512 = 4 x 32 tiles, K=128
  for (int t = wave; t < 128; t += 8) {
    int mt = t >> 5, nt = t & 31;
    f32x8 acc = {};
    const bf16* Bb = fc1_wT + nt * 16 * CDIM;
#pragma unroll
    for (int ks = 0; ks < 4; ++ks) {
      if (ks < 3) pf(Bb + (ks + 1) * 32 + (lane & 15) * CDIM);
      acc = wmma_bf16(load_fragA(Ah + mt * 16 * CDIM + ks * 32, CDIM),
                      load_fragB(Bb + ks * 32, CDIM), acc);
    }
    int col = nt * 16 + (lane & 15);
    int rb = (lane >> 4) * 8;
    float fb = fc1_b[col];
#pragma unroll
    for (int i = 0; i < 8; ++i) {
      int m = mt * 16 + rb + i;
      float v = acc[i] + fb;
      v = 0.5f * v * (1.f + erff(v * 0.70710678118654752f));  // exact GELU
      hmlp[(tok0 + m) * MLPD + col] = (bf16)v;
    }
  }
}

// ---------------- K5: fc2 + residual ----------------
__global__ void __launch_bounds__(256)
k5_fc2(const bf16* __restrict__ hmlp, const bf16* __restrict__ fc2_wT,
       const float* __restrict__ fc2_b, float* __restrict__ out) {
  size_t tok0 = (size_t)blockIdx.x * NP;
  int lane = lane_id(), wave = wave_id();
  for (int t = wave; t < 32; t += 8) {
    int mt = t >> 3, nt = t & 7;
    f32x8 acc = {};
    const bf16* Ab = hmlp + (tok0 + mt * 16) * MLPD;
    const bf16* Bb = fc2_wT + nt * 16 * MLPD;
#pragma unroll
    for (int ks = 0; ks < 16; ++ks) {
      if (ks < 15) {
        pf(Ab + (ks + 1) * 32 + (lane & 15) * MLPD);
        pf(Bb + (ks + 1) * 32 + (lane & 15) * MLPD);
      }
      acc = wmma_bf16(load_fragA(Ab + ks * 32, MLPD), load_fragB(Bb + ks * 32, MLPD), acc);
    }
    int col = nt * 16 + (lane & 15);
    int rb = (lane >> 4) * 8;
    float fb = fc2_b[col];
#pragma unroll
    for (int i = 0; i < 8; ++i) {
      size_t pos = (tok0 + mt * 16 + rb + i) * CDIM + col;
      out[pos] = out[pos] + acc[i] + fb;   // += x1 stored by K3
    }
  }
}

extern "C" void kernel_launch(void* const* d_in, const int* in_sizes, int n_in,
                              void* d_out, int out_size, void* d_ws, size_t ws_size,
                              hipStream_t stream) {
  const float* x      = (const float*)d_in[0];
  const float* qkv_w  = (const float*)d_in[1];
  const float* qkv_b  = (const float*)d_in[2];
  const float* proj_w = (const float*)d_in[3];
  const float* proj_b = (const float*)d_in[4];
  const float* reltab = (const float*)d_in[5];
  const float* g1     = (const float*)d_in[6];
  const float* b1     = (const float*)d_in[7];
  const float* g2     = (const float*)d_in[8];
  const float* b2     = (const float*)d_in[9];
  const float* fc1_w  = (const float*)d_in[10];
  const float* fc1_b  = (const float*)d_in[11];
  const float* fc2_w  = (const float*)d_in[12];
  const float* fc2_b  = (const float*)d_in[13];
  float* out = (float*)d_out;

  char* ws = (char*)d_ws;
  size_t off = 0;
  auto take = [&](size_t bytes) -> char* {
    char* p = ws + off;
    off = (off + bytes + 255) & ~(size_t)255;
    return p;
  };
  bf16* qkv_wT  = (bf16*)take((size_t)384 * 128 * sizeof(bf16));
  bf16* proj_wT = (bf16*)take((size_t)128 * 128 * sizeof(bf16));
  bf16* fc1_wT  = (bf16*)take((size_t)512 * 128 * sizeof(bf16));
  bf16* fc2_wT  = (bf16*)take((size_t)128 * 512 * sizeof(bf16));
  float* biasT  = (float*)take((size_t)HEADS * NP * NP * sizeof(float));
  float* maskT  = (float*)take((size_t)NWIN * NP * NP * sizeof(float));
  size_t attn_off = off;
  bf16* qbuf = (bf16*)take((size_t)BW * HEADS * NP * HDIM * sizeof(bf16));
  bf16* kbuf = (bf16*)take((size_t)BW * HEADS * NP * HDIM * sizeof(bf16));
  bf16* vT   = (bf16*)take((size_t)BW * HEADS * HDIM * NP * sizeof(bf16));
  bf16* owin = (bf16*)take((size_t)BW * NP * CDIM * sizeof(bf16));
  // MLP hidden (100352*512 bf16 = 103 MB) aliases q/k/v/owin (134 MB, dead by K4)
  bf16* hmlp = (bf16*)(ws + attn_off);

  hipLaunchKernelGGL(prep_transpose, dim3(192), dim3(256), 0, stream, qkv_w, qkv_wT, 128, 384);
  hipLaunchKernelGGL(prep_transpose, dim3(64),  dim3(256), 0, stream, proj_w, proj_wT, 128, 128);
  hipLaunchKernelGGL(prep_transpose, dim3(256), dim3(256), 0, stream, fc1_w, fc1_wT, 128, 512);
  hipLaunchKernelGGL(prep_transpose, dim3(256), dim3(256), 0, stream, fc2_w, fc2_wT, 512, 128);
  hipLaunchKernelGGL(prep_bias_mask, dim3(512), dim3(256), 0, stream, reltab, biasT, maskT);

  hipLaunchKernelGGL(k1_ln_qkv, dim3(BW), dim3(256), 0, stream,
                     x, g1, b1, qkv_wT, qkv_b, qbuf, kbuf, vT);
  hipLaunchKernelGGL(k2_attn, dim3(BW * HEADS), dim3(256), 0, stream,
                     qbuf, kbuf, vT, biasT, maskT, owin);
  hipLaunchKernelGGL(k3_proj, dim3(BW), dim3(256), 0, stream,
                     x, owin, proj_wT, proj_b, out);
  hipLaunchKernelGGL(k4_ln_fc1, dim3(NTOK / NP), dim3(256), 0, stream,
                     out, g2, b2, fc1_wT, fc1_b, hmlp);
  hipLaunchKernelGGL(k5_fc2, dim3(NTOK / NP), dim3(256), 0, stream,
                     hmlp, fc2_wT, fc2_b, out);
}
